// SelfAttention_48344151883815
// MI455X (gfx1250) — compile-verified, matrix operand order checked by reading, not run on hardware
//
#include <hip/hip_runtime.h>

// ---------------------------------------------------------------------------
// Self-attention block for MI455X (gfx1250, wave32, WMMA).
//   x[2048,3072] @ Wqkv -> qkv ; RMSNorm+RoPE ; flash attention (24 heads,
//   d=128, S=2048) ; @ Wo -> out.  All GEMM-like math via
//   v_wmma_f32_16x16x32_bf16 (bf16 in, f32 accumulate).  Tile staging uses
//   GLOBAL_LOAD_ASYNC_TO_LDS_B128 (ASYNCcnt) when the builtin is available.
// ---------------------------------------------------------------------------

#define S_LEN 2048
#define HID   3072
#define NH    24
#define HD    128
#define QKV_N (3 * HID)

typedef __attribute__((ext_vector_type(16))) __bf16 v16bf;
typedef __attribute__((ext_vector_type(8)))  float  v8f;
typedef __attribute__((ext_vector_type(4)))  int    i32x4;

union B8 { uint4 q; __bf16 h[8]; };
union B4 { uint2 u; __bf16 h[4]; };

#define AS1 __attribute__((address_space(1)))
#define AS3 __attribute__((address_space(3)))

#if defined(__has_builtin)
#if __has_builtin(__builtin_amdgcn_global_load_async_to_lds_b128)
#define HAVE_ASYNC_LDS 1
#endif
#endif
#ifndef HAVE_ASYNC_LDS
#define HAVE_ASYNC_LDS 0
#endif

// 16-byte global -> LDS copy.  Async (ASYNCcnt-tracked, no VGPR round trip)
// when available; plain load+ds_store otherwise.  The builtin takes typed
// int4 pointers in AS1/AS3; build them via integer round-trips (generic LDS
// pointers carry the LDS offset in their low 32 bits).
__device__ __forceinline__ void copy16_g2lds(const __bf16* gsrc, __bf16* ldst) {
#if HAVE_ASYNC_LDS
  __builtin_amdgcn_global_load_async_to_lds_b128(
      (AS1 i32x4*)(unsigned long long)gsrc,
      (AS3 i32x4*)(unsigned int)(unsigned long long)ldst, 0, 0);
#else
  *(uint4*)ldst = *(const uint4*)gsrc;
#endif
}

__device__ __forceinline__ void wait_async_lds() {
#if HAVE_ASYNC_LDS
#if __has_builtin(__builtin_amdgcn_s_wait_asynccnt)
  __builtin_amdgcn_s_wait_asynccnt(0);
#else
  asm volatile("s_wait_asynccnt 0" ::: "memory");
#endif
#endif
}

__device__ __forceinline__ v8f wmma_bf16(v16bf a, v16bf b, v8f c) {
  // D = A(16x32 bf16) * B(32x16 bf16) + C(16x16 f32)
  return __builtin_amdgcn_wmma_f32_16x16x32_bf16(
      /*neg_a=*/false, a, /*neg_b=*/false, b,
      /*c_mod=*/(short)0, c, /*reuse_a=*/false, /*reuse_b=*/false);
}

// Load one 16x32 (or 32x16) bf16 WMMA fragment from an LDS row laid out
// [M (or N)][K] with K contiguous.  Per the CDNA5 16-bit A-matrix layout:
// lanes 0-15 hold K = {0..7, 16..23}, lanes 16-31 hold K = {8..15, 24..31}.
__device__ __forceinline__ v16bf ld_frag(const __bf16* p, int hs) {
  v16bf f;
  ((uint4*)&f)[0] = *(const uint4*)(p + hs * 8);        // K block 0/1
  ((uint4*)&f)[1] = *(const uint4*)(p + 16 + hs * 8);   // K block 2/3
  return f;
}

// ---------------------------------------------------------------------------
// f32 -> bf16 conversion (vectorized: float4 in, 4x bf16 packed out)
// ---------------------------------------------------------------------------
__global__ void cvt_f32_bf16(const float* __restrict__ in,
                             __bf16* __restrict__ out, long n4) {
  long i = (long)blockIdx.x * blockDim.x + threadIdx.x;
  if (i < n4) {
    float4 v = ((const float4*)in)[i];
    B4 b;
    b.h[0] = (__bf16)v.x; b.h[1] = (__bf16)v.y;
    b.h[2] = (__bf16)v.z; b.h[3] = (__bf16)v.w;
    ((uint2*)out)[i] = b.u;
  }
}

// ---------------------------------------------------------------------------
// Tiled bf16 GEMM: C[M,N](f32) = A[M,K](bf16) @ B[K,N](bf16)
// 128x128 tile per block, BK=32, 256 threads = 8 waves, wave -> 32x64,
// double-buffered LDS.  A tile staged with async global->LDS; B tile is
// transposed into LDS ([n][k]) so fragments are contiguous ds_load_b128s.
// ---------------------------------------------------------------------------
__global__ __launch_bounds__(256) void gemm_bf16_kernel(
    const __bf16* __restrict__ A, const __bf16* __restrict__ B,
    float* __restrict__ C, int M, int N, int K) {
  __shared__ alignas(16) __bf16 As[2][128][40];  // [m][k], stride 80B
  __shared__ alignas(16) __bf16 Bs[2][128][40];  // [n][k] (transposed)

  const int tid = threadIdx.x;
  const int w = tid >> 5, lane = tid & 31;
  const int lh = lane & 15, hs = lane >> 4;
  const int wm = w >> 1, wn = w & 1;           // wave grid 4x2
  const int m0 = blockIdx.y * 128, n0 = blockIdx.x * 128;

  v8f acc[2][4];
  v8f zero = {};
#pragma unroll
  for (int a = 0; a < 2; ++a)
#pragma unroll
    for (int b = 0; b < 4; ++b) acc[a][b] = zero;

  auto load_tiles = [&](int kt, int buf) {
    const int k0 = kt * 32;
    // A tile: 128x32, async global->LDS, two 16B transfers per thread
#pragma unroll
    for (int p = 0; p < 2; ++p) {
      int r = (tid >> 2) + p * 64;
      int kc = (tid & 3) * 8;
      copy16_g2lds(A + (size_t)(m0 + r) * K + k0 + kc, &As[buf][r][kc]);
    }
    // B tile: 32x128, loaded row-major, stored transposed [n][k]
#pragma unroll
    for (int p = 0; p < 2; ++p) {
      int u = tid + p * 256;
      int kk = u >> 4;
      int nc = (u & 15) * 8;
      B8 un;
      un.q = *(const uint4*)(B + (size_t)(k0 + kk) * N + n0 + nc);
#pragma unroll
      for (int i = 0; i < 8; ++i) Bs[buf][nc + i][kk] = un.h[i];
    }
  };

  const int nkt = K / 32;
  load_tiles(0, 0);
  wait_async_lds();
  __syncthreads();

  for (int kt = 0; kt < nkt; ++kt) {
    const int buf = kt & 1;
    if (kt + 2 < nkt) {  // L2 prefetch (global_prefetch_b8)
      __builtin_prefetch(A + (size_t)(m0 + (tid >> 2)) * K + (kt + 2) * 32, 0, 1);
      __builtin_prefetch(B + (size_t)((kt + 2) * 32 + (tid >> 4)) * N + n0, 0, 1);
    }
    if (kt + 1 < nkt) load_tiles(kt + 1, buf ^ 1);

    // Register-blocked fragments: one dscnt wait covers the 8-WMMA burst.
    v16bf af[2], bfv[4];
#pragma unroll
    for (int tm = 0; tm < 2; ++tm)
      af[tm] = ld_frag(&As[buf][wm * 32 + tm * 16 + lh][0], hs);
#pragma unroll
    for (int tn = 0; tn < 4; ++tn)
      bfv[tn] = ld_frag(&Bs[buf][wn * 64 + tn * 16 + lh][0], hs);
#pragma unroll
    for (int tn = 0; tn < 4; ++tn)
#pragma unroll
      for (int tm = 0; tm < 2; ++tm)
        acc[tm][tn] = wmma_bf16(af[tm], bfv[tn], acc[tm][tn]);

    wait_async_lds();  // next tile's async copies done before barrier
    __syncthreads();
  }

  // C/D layout: VGPR i holds row (i + 8*hs), column = lane%16.
#pragma unroll
  for (int tm = 0; tm < 2; ++tm)
#pragma unroll
    for (int tn = 0; tn < 4; ++tn)
#pragma unroll
      for (int i = 0; i < 8; ++i) {
        int row = m0 + wm * 32 + tm * 16 + hs * 8 + i;
        int col = n0 + wn * 64 + tn * 16 + lh;
        C[(size_t)row * N + col] = acc[tm][tn][i];
      }
}

// ---------------------------------------------------------------------------
// RMSNorm + 3-axis RoPE.  One 128-thread block per (seq, head).
// qkv f32 [S, 24, 384] -> q/k/v bf16 [head][seq][128]
// ---------------------------------------------------------------------------
__global__ __launch_bounds__(128) void normrope_kernel(
    const float* __restrict__ qkv, const float* __restrict__ qw,
    const float* __restrict__ kw, const float* __restrict__ cosb,
    const float* __restrict__ sinb, __bf16* __restrict__ qo,
    __bf16* __restrict__ ko, __bf16* __restrict__ vo) {
  const int h = blockIdx.x, s = blockIdx.y, d = threadIdx.x;
  const size_t base = (size_t)s * QKV_N + (size_t)h * (3 * HD);
  float qv = qkv[base + d];
  float kv = qkv[base + HD + d];
  float vv = qkv[base + 2 * HD + d];

  __shared__ float red[128];
  __shared__ float qb[128], kb2[128];

  // RMS(q)
  red[d] = qv * qv;
  __syncthreads();
#pragma unroll
  for (int st = 64; st > 0; st >>= 1) {
    if (d < st) red[d] += red[d + st];
    __syncthreads();
  }
  float qn = qv * rsqrtf(red[0] * (1.0f / HD) + 1e-6f) * qw[d];
  __syncthreads();
  // RMS(k)
  red[d] = kv * kv;
  __syncthreads();
#pragma unroll
  for (int st = 64; st > 0; st >>= 1) {
    if (d < st) red[d] += red[d + st];
    __syncthreads();
  }
  float kn = kv * rsqrtf(red[0] * (1.0f / HD) + 1e-6f) * kw[d];
  __syncthreads();

  // RoPE over splits (64,32,32); cos/sin row width = 32+16+16 = 64
  qb[d] = qn;
  kb2[d] = kn;
  __syncthreads();

  int o, hlf, coff;
  if (d < 64)      { o = 0;  hlf = 32; coff = 0;  }
  else if (d < 96) { o = 64; hlf = 16; coff = 32; }
  else             { o = 96; hlf = 16; coff = 48; }
  const int j = d - o;
  const float* cs = cosb + (size_t)s * 64;
  const float* sn = sinb + (size_t)s * 64;
  float qr, kr;
  if (j < hlf) {
    float c1 = cs[coff + j], s1 = sn[coff + j];
    qr = qb[o + j] * c1 - qb[o + hlf + j] * s1;
    kr = kb2[o + j] * c1 - kb2[o + hlf + j] * s1;
  } else {
    int j2 = j - hlf;
    float c1 = cs[coff + j2], s1 = sn[coff + j2];
    qr = qb[o + hlf + j2] * c1 + qb[o + j2] * s1;
    kr = kb2[o + hlf + j2] * c1 + kb2[o + j2] * s1;
  }

  const size_t oidx = ((size_t)h * S_LEN + s) * HD + d;
  qo[oidx] = (__bf16)qr;
  ko[oidx] = (__bf16)kr;
  vo[oidx] = (__bf16)vv;
}

// ---------------------------------------------------------------------------
// Flash attention, one block per (128-row query tile, head).
// 256 threads = 8 waves; wave w owns query rows [16w, 16w+16).
// Q tile A-fragments live in registers across the whole key loop.
// Online softmax on the 16x16 C layout (row = vgpr + 8*hs, col = lane%16).
// ---------------------------------------------------------------------------
__global__ __launch_bounds__(256) void attn_kernel(
    const __bf16* __restrict__ Q, const __bf16* __restrict__ Kg,
    const __bf16* __restrict__ V, __bf16* __restrict__ O) {
  __shared__ alignas(16) __bf16 sQP[128][136];  // Q staging, then P buffer
  __shared__ alignas(16) __bf16 sK[128][136];   // [key][d]
  __shared__ alignas(16) __bf16 sVt[128][136];  // [d][key] (transposed)

  const int tid = threadIdx.x;
  const int w = tid >> 5, lane = tid & 31;
  const int lh = lane & 15, hs = lane >> 4;
  const int h = blockIdx.y, q0 = blockIdx.x * 128;

  const __bf16* Qh = Q + (size_t)h * S_LEN * HD;
  const __bf16* Kh = Kg + (size_t)h * S_LEN * HD;
  const __bf16* Vh = V + (size_t)h * S_LEN * HD;

  // Stage Q tile (128x128 bf16) async, then pull per-wave A-frags to regs.
#pragma unroll
  for (int p = 0; p < 8; ++p) {
    int u = tid + p * 256, r = u >> 4, c = (u & 15) * 8;
    copy16_g2lds(Qh + (size_t)(q0 + r) * HD + c, &sQP[r][c]);
  }
  wait_async_lds();
  __syncthreads();
  v16bf qf[4];
#pragma unroll
  for (int kb = 0; kb < 4; ++kb)
    qf[kb] = ld_frag(&sQP[w * 16 + lh][kb * 32], hs);
  asm volatile("s_wait_dscnt 0" ::: "memory");  // qf resident before sQP reuse
  __syncthreads();

  float mrow[8], lrow[8];
  v8f o_acc[8];
  v8f zero = {};
#pragma unroll
  for (int i = 0; i < 8; ++i) { mrow[i] = -3.0e38f; lrow[i] = 0.f; o_acc[i] = zero; }

  const float scale = 0.08838834764831845f;  // 1/sqrt(128)

  for (int kt = 0; kt < S_LEN / 128; ++kt) {
    const int k0 = kt * 128;
    // K tile natural [key][d] (async copy); V tile transposed [d][key].
#pragma unroll
    for (int p = 0; p < 8; ++p) {
      int u = tid + p * 256, r = u >> 4, c = (u & 15) * 8;
      copy16_g2lds(Kh + (size_t)(k0 + r) * HD + c, &sK[r][c]);
      B8 un;
      un.q = *(const uint4*)(Vh + (size_t)(k0 + r) * HD + c);
#pragma unroll
      for (int i = 0; i < 8; ++i) sVt[c + i][r] = un.h[i];
    }
    wait_async_lds();
    __syncthreads();

    // S = Q @ K^T for this wave's 16 rows x 128 keys
    v8f s_acc[8];
#pragma unroll
    for (int tn = 0; tn < 8; ++tn) {
      v16bf kf[4];
#pragma unroll
      for (int kb = 0; kb < 4; ++kb)
        kf[kb] = ld_frag(&sK[tn * 16 + lh][kb * 32], hs);
      s_acc[tn] = zero;
#pragma unroll
      for (int kb = 0; kb < 4; ++kb)
        s_acc[tn] = wmma_bf16(qf[kb], kf[kb], s_acc[tn]);
    }

    // Online softmax (row-wise; cols live across the 16-lane half-wave).
    float mnew[8], corr[8];
#pragma unroll
    for (int i = 0; i < 8; ++i) {
      float m = -3.0e38f;
#pragma unroll
      for (int tn = 0; tn < 8; ++tn) {
        s_acc[tn][i] *= scale;
        m = fmaxf(m, s_acc[tn][i]);
      }
      m = fmaxf(m, __shfl_xor(m, 1, 16));
      m = fmaxf(m, __shfl_xor(m, 2, 16));
      m = fmaxf(m, __shfl_xor(m, 4, 16));
      m = fmaxf(m, __shfl_xor(m, 8, 16));
      mnew[i] = fmaxf(mrow[i], m);
      corr[i] = __expf(mrow[i] - mnew[i]);
      mrow[i] = mnew[i];
      float rs = 0.f;
#pragma unroll
      for (int tn = 0; tn < 8; ++tn) {
        float pv = __expf(s_acc[tn][i] - mnew[i]);
        s_acc[tn][i] = pv;
        rs += pv;
      }
      rs += __shfl_xor(rs, 1, 16);
      rs += __shfl_xor(rs, 2, 16);
      rs += __shfl_xor(rs, 4, 16);
      rs += __shfl_xor(rs, 8, 16);
      lrow[i] = lrow[i] * corr[i] + rs;
#pragma unroll
      for (int tn = 0; tn < 8; ++tn) o_acc[tn][i] *= corr[i];
    }

    // P (C layout) -> LDS (wave-private rows of sQP) to re-shape as A frags.
#pragma unroll
    for (int tn = 0; tn < 8; ++tn)
#pragma unroll
      for (int i = 0; i < 8; ++i)
        sQP[w * 16 + hs * 8 + i][tn * 16 + lh] = (__bf16)s_acc[tn][i];

    // O += P @ V
    v16bf pf[4];
#pragma unroll
    for (int kb = 0; kb < 4; ++kb)
      pf[kb] = ld_frag(&sQP[w * 16 + lh][kb * 32], hs);
#pragma unroll
    for (int tn = 0; tn < 8; ++tn) {
      v16bf vf[4];
#pragma unroll
      for (int kb = 0; kb < 4; ++kb)
        vf[kb] = ld_frag(&sVt[tn * 16 + lh][kb * 32], hs);
#pragma unroll
      for (int kb = 0; kb < 4; ++kb)
        o_acc[tn] = wmma_bf16(pf[kb], vf[kb], o_acc[tn]);
    }
    __syncthreads();
  }

  // Normalize and emit bf16 in [s][h*128+d] layout for the output GEMM.
  float invl[8];
#pragma unroll
  for (int i = 0; i < 8; ++i) invl[i] = 1.0f / lrow[i];
#pragma unroll
  for (int tn = 0; tn < 8; ++tn)
#pragma unroll
    for (int i = 0; i < 8; ++i) {
      int row = q0 + w * 16 + hs * 8 + i;
      int col = tn * 16 + lh;
      O[(size_t)row * HID + h * HD + col] = (__bf16)(o_acc[tn][i] * invl[i]);
    }
}

// ---------------------------------------------------------------------------
// Host orchestration
// ---------------------------------------------------------------------------
extern "C" void kernel_launch(void* const* d_in, const int* in_sizes, int n_in,
                              void* d_out, int out_size, void* d_ws,
                              size_t ws_size, hipStream_t stream) {
  const float* x    = (const float*)d_in[0];
  const float* Wqkv = (const float*)d_in[1];
  const float* Wo   = (const float*)d_in[2];
  const float* qw   = (const float*)d_in[3];
  const float* kw   = (const float*)d_in[4];
  const float* cosb = (const float*)d_in[5];
  const float* sinb = (const float*)d_in[6];
  float* out = (float*)d_out;

  char* p = (char*)d_ws;
  auto alloc = [&](size_t bytes) -> void* {
    void* r = (void*)p;
    p += (bytes + 255) & ~(size_t)255;
    return r;
  };
  const size_t n_x    = (size_t)S_LEN * HID;
  const size_t n_wqkv = (size_t)HID * QKV_N;
  const size_t n_wo   = (size_t)HID * HID;
  const size_t n_head = (size_t)NH * S_LEN * HD;

  __bf16* xb    = (__bf16*)alloc(n_x * 2);
  __bf16* wqkvb = (__bf16*)alloc(n_wqkv * 2);
  __bf16* wob   = (__bf16*)alloc(n_wo * 2);
  float*  qkvf  = (float*)alloc((size_t)S_LEN * QKV_N * 4);
  __bf16* qh    = (__bf16*)alloc(n_head * 2);
  __bf16* kh    = (__bf16*)alloc(n_head * 2);
  __bf16* vh    = (__bf16*)alloc(n_head * 2);
  __bf16* oh    = (__bf16*)alloc(n_x * 2);

  // 1) f32 -> bf16 conversions
  cvt_f32_bf16<<<dim3((unsigned)((n_x / 4 + 255) / 256)), 256, 0, stream>>>(
      x, xb, (long)(n_x / 4));
  cvt_f32_bf16<<<dim3((unsigned)((n_wqkv / 4 + 255) / 256)), 256, 0, stream>>>(
      Wqkv, wqkvb, (long)(n_wqkv / 4));
  cvt_f32_bf16<<<dim3((unsigned)((n_wo / 4 + 255) / 256)), 256, 0, stream>>>(
      Wo, wob, (long)(n_wo / 4));

  // 2) QKV projection: [2048,3072] @ [3072,9216] -> f32
  gemm_bf16_kernel<<<dim3(QKV_N / 128, S_LEN / 128), 256, 0, stream>>>(
      xb, wqkvb, qkvf, S_LEN, QKV_N, HID);

  // 3) RMSNorm + RoPE, split heads to bf16
  normrope_kernel<<<dim3(NH, S_LEN), 128, 0, stream>>>(
      qkvf, qw, kw, cosb, sinb, qh, kh, vh);

  // 4) Flash attention
  attn_kernel<<<dim3(S_LEN / 128, NH), 256, 0, stream>>>(qh, kh, vh, oh);

  // 5) Output projection: [2048,3072] @ [3072,3072] -> f32 out
  gemm_bf16_kernel<<<dim3(HID / 128, S_LEN / 128), 256, 0, stream>>>(
      oh, wob, out, S_LEN, HID, HID);
}
